// LSTM_43379169690067
// MI455X (gfx1250) — matrix operand order repaired and checked by reading, not compiled, not run on hardware
//
#include <hip/hip_runtime.h>

// ---------------------------------------------------------------------------
// LSTM (BATCH=256, SEQ=256, IN=1024, HID=1024) for gfx1250 (MI455X, wave32).
//
//  * x-projections hoisted into one big bf16 WMMA GEMM (half the FLOPs,
//    fully parallel); recurrent h@Wh GEMM runs 256x sequentially with
//    Wh (8MB bf16) L2-resident.
//  * A operands (X, h) pre-packed into WMMA A-fragment tiles -> direct
//    global loads, no LDS, no shuffles in the hot loop.
//  * B operands (Wi, Wh) pre-packed into WMMA B-fragment tiles -> LDS
//    staging is a bulk copy via global_load_async_to_lds_b128 (ASYNCcnt),
//    double-buffered to overlap with v_wmma_f32_16x16x32_bf16 compute.
//  * Big GEMM: 2 A-tiles per wave -> 8 WMMA per k-tile per wave against
//    4 B-frag LDS loads (B reused across both A tiles).
// ---------------------------------------------------------------------------

#define BATCH 256
#define SEQ   256
#define INF_  1024
#define HID   1024
#define G4    (4 * HID)        // 4096
#define BS    (BATCH * SEQ)    // 65536
#define KT    32               // K tiles of 32 (K = 1024 for both GEMMs)
#define NTILES_B (G4 / 16)     // 256 B n-tiles per k-tile row

typedef __attribute__((ext_vector_type(16))) __bf16 v16bf;
typedef __attribute__((ext_vector_type(8)))  float  v8f;

// ---------------- bf16 helpers --------------------------------------------
static __device__ __forceinline__ unsigned short f2bf(float f) {
    union { float f; unsigned u; } v; v.f = f;
    unsigned r = v.u + 0x7FFFu + ((v.u >> 16) & 1u);   // round-to-nearest-even
    return (unsigned short)(r >> 16);
}
static __device__ __forceinline__ float bf2f(unsigned short s) {
    union { unsigned u; float f; } v; v.u = ((unsigned)s) << 16;
    return v.f;
}

union FragU { uint4 q[2]; v16bf v; };

// frag = 32 contiguous bytes at p (16 halves), matching pre-packed layouts
static __device__ __forceinline__ v16bf load_frag(const unsigned short* p) {
    FragU u;
    u.q[0] = *reinterpret_cast<const uint4*>(p);
    u.q[1] = *reinterpret_cast<const uint4*>(p + 8);
    return u.v;
}

// ---------------- CDNA5 async global->LDS copy (ASYNCcnt) ------------------
static __device__ __forceinline__ void async_b128(unsigned lds_byte_addr,
                                                  const void* gsrc) {
    unsigned long long ga = (unsigned long long)gsrc;
    asm volatile("global_load_async_to_lds_b128 %0, %1, off"
                 :: "v"(lds_byte_addr), "v"(ga) : "memory");
}
static __device__ __forceinline__ void wait_async_le1() {
    asm volatile("s_wait_asynccnt 0x1" ::: "memory");
}
static __device__ __forceinline__ void wait_async_0() {
    asm volatile("s_wait_asynccnt 0x0" ::: "memory");
}

// ---------------- WMMA tile GEMM core --------------------------------------
// Block = 256 threads (8 waves). Each wave: AT A-tiles (AT*16 rows) x 64 cols.
// A: packed frag tiles in global, loaded directly per wave.
// B: packed frag tiles in global, async-copied to double-buffered LDS.
//    Per k-tile the block stages 4 n-tiles * 512 halves = 4KB (1 b128/thread).
template <int AT>
static __device__ __forceinline__ void gemm_core(
    const unsigned short* __restrict__ Apack, int tmWave0,
    const unsigned short* __restrict__ Bpack, int tn0,
    unsigned short* ldsB /* 2 x 2048 halves */, v8f* acc /* [AT*4] */)
{
    const int tid  = threadIdx.x;
    const int lane = tid & 31;

    const unsigned ldsBase =
        (unsigned)(unsigned long long)(const void*)ldsB + (unsigned)tid * 16u;

    // prologue: stage k-tile 0 into buffer 0
    async_b128(ldsBase, Bpack + ((size_t)tn0) * 512 + tid * 8);

    for (int tk = 0; tk < KT; ++tk) {
        __syncthreads();                       // everyone done with other buf
        if (tk + 1 < KT) {
            async_b128(ldsBase + (unsigned)((tk + 1) & 1) * 4096u,
                       Bpack + ((size_t)((tk + 1) * NTILES_B + tn0)) * 512 + tid * 8);
            wait_async_le1();                  // k-tile tk finished landing
        } else {
            wait_async_0();
        }
        __syncthreads();                       // stage visible to all waves

        v16bf af[AT];
#pragma unroll
        for (int at = 0; at < AT; ++at) {
            const unsigned short* ap =
                Apack + ((size_t)((tmWave0 + at) * KT + tk)) * 512 + lane * 16;
            if (tk + 1 < KT)
                __builtin_prefetch(ap + 512, 0, 1);
            af[at] = load_frag(ap);
        }

        const unsigned short* buf = ldsB + (tk & 1) * 2048;
#pragma unroll
        for (int nt = 0; nt < 4; ++nt) {
            v16bf bf = load_frag(buf + nt * 512 + lane * 16);
#pragma unroll
            for (int at = 0; at < AT; ++at) {
                acc[at * 4 + nt] = __builtin_amdgcn_wmma_f32_16x16x32_bf16(
                    false, af[at], false, bf, (short)0, acc[at * 4 + nt],
                    false, false);
            }
        }
    }
}

// ---------------- kernel: big x-projection GEMM ----------------------------
// A rows are m' = t*BATCH + b (t-major) so C stores land contiguously in the
// gates_x[t][b][4H] layout the recurrent step consumes. 2 A-tiles per wave.
__global__ __launch_bounds__(256)
void lstm_gemm_x(const unsigned short* __restrict__ Xpack,
                 const unsigned short* __restrict__ WiP,
                 unsigned short* __restrict__ gatesx)
{
    __shared__ unsigned short ldsB[2 * 2048];
    const int mBase = blockIdx.x * 256;      // over 65536
    const int nBase = blockIdx.y * 64;       // over 4096
    const int wave  = threadIdx.x >> 5;
    const int lane  = threadIdx.x & 31;
    v8f acc[8] = {};
    gemm_core<2>(Xpack, (mBase >> 4) + wave * 2, WiP, nBase >> 4, ldsB, acc);

    const int hi = lane >> 4, lm = lane & 15;
#pragma unroll
    for (int at = 0; at < 2; ++at) {
#pragma unroll
        for (int nt = 0; nt < 4; ++nt) {
#pragma unroll
            for (int r = 0; r < 8; ++r) {
                int m = mBase + (wave * 2 + at) * 16 + r + hi * 8; // t*BATCH+b
                int n = nBase + nt * 16 + lm;
                gatesx[(size_t)m * G4 + n] = f2bf(acc[at * 4 + nt][r]);
            }
        }
    }
}

// ---------------- kernel: recurrent GEMM (one step) ------------------------
__global__ __launch_bounds__(256)
void lstm_gemm_h(const unsigned short* __restrict__ Hpack,
                 const unsigned short* __restrict__ WhP,
                 const unsigned short* __restrict__ gx_t,
                 float* __restrict__ gates)
{
    __shared__ unsigned short ldsB[2 * 2048];
    const int mBase = blockIdx.x * 128;      // over 256
    const int nBase = blockIdx.y * 64;       // over 4096
    const int wave  = threadIdx.x >> 5;
    const int lane  = threadIdx.x & 31;
    v8f acc[4] = {};
    gemm_core<1>(Hpack, (mBase >> 4) + wave, WhP, nBase >> 4, ldsB, acc);

    const int hi = lane >> 4, lm = lane & 15;
#pragma unroll
    for (int nt = 0; nt < 4; ++nt) {
#pragma unroll
        for (int r = 0; r < 8; ++r) {
            int m = mBase + wave * 16 + r + hi * 8;    // batch index
            int n = nBase + nt * 16 + lm;
            size_t idx = (size_t)m * G4 + n;
            gates[idx] = acc[nt][r] + bf2f(gx_t[idx]);
        }
    }
}

// ---------------- kernel: elementwise cell update --------------------------
__global__ __launch_bounds__(256)
void lstm_cell(const float* __restrict__ gates, float* __restrict__ c,
               unsigned short* __restrict__ Hpack,
               const float* __restrict__ bii, const float* __restrict__ bhi,
               const float* __restrict__ bif, const float* __restrict__ bhf,
               const float* __restrict__ big, const float* __restrict__ bhg,
               const float* __restrict__ bio, const float* __restrict__ bho,
               float* __restrict__ out, int write_out)
{
    int idx = blockIdx.x * 256 + threadIdx.x;      // 0 .. BATCH*HID-1
    int b = idx >> 10, j = idx & 1023;
    const float* g = gates + (size_t)b * G4;
    float xi = g[j]        + bii[j] + bhi[j];
    float xf = g[1024 + j] + bif[j] + bhf[j];
    float xg = g[2048 + j] + big[j] + bhg[j];
    float xo = g[3072 + j] + bio[j] + bho[j];
    float i_t = 1.0f / (1.0f + __expf(-xi));
    float f_t = 1.0f / (1.0f + __expf(-xf));
    float g_t = tanhf(xg);
    float o_t = 1.0f / (1.0f + __expf(-xo));
    float ct = f_t * c[idx] + i_t * g_t;
    float ht = o_t * tanhf(ct);
    c[idx] = ct;
    // write h in A-fragment packed layout for next step's GEMM
    {
        int tile = (b >> 4) * KT + (j >> 5);
        int lA   = (b & 15) + 16 * ((j >> 3) & 1);
        int pA   = (j & 7) + 8 * ((j >> 4) & 1);
        Hpack[(size_t)tile * 512 + lA * 16 + pA] = f2bf(ht);
    }
    if (write_out) {
        out[idx] = ht;                      // h_t first
        out[BATCH * HID + idx] = ct;        // then c_t
    }
}

// ---------------- pack kernels (dest-indexed, coalesced writes) ------------
// X -> A-fragment tiles, rows reordered to m' = t*BATCH + b.
__global__ __launch_bounds__(256)
void pack_x(const float* __restrict__ x, unsigned short* __restrict__ Xpack) {
    size_t i = (size_t)blockIdx.x * 256 + threadIdx.x;   // BS*IN dest elems
    int tile = (int)(i >> 9);
    int lA   = (int)(i >> 4) & 31;
    int pos  = (int)i & 15;
    int tm = tile >> 5;            // m-tile (0..4095)
    int tk = tile & 31;            // k-tile (0..31)
    int mp = tm * 16 + (lA & 15);  // m' = t*BATCH + b
    int k  = tk * 32 + ((lA >> 4) & 1) * 8 + (pos & 7) + ((pos >> 3) & 1) * 16;
    int b = mp & 255, t = mp >> 8;
    Xpack[i] = f2bf(x[((size_t)b * SEQ + t) * INF_ + k]);
}

// 4 weight matrices -> concatenated B-fragment tiles.
// B layout: lane = (n&15) | ((k>>3)&1)<<4, elem = (k&7) | ((k>>4)&1)<<3.
__global__ __launch_bounds__(256)
void pack_w(const float* __restrict__ wi, const float* __restrict__ wf,
            const float* __restrict__ wg, const float* __restrict__ wo,
            unsigned short* __restrict__ Bpack) {
    size_t i = (size_t)blockIdx.x * 256 + threadIdx.x;   // 1024*4096 dest elems
    int tile = (int)(i >> 9);
    int lB   = (int)(i >> 4) & 31;
    int pos  = (int)i & 15;
    int tk = tile >> 8;            // k-tile (0..31)
    int tn = tile & 255;           // n-tile (0..255)
    int n = tn * 16 + (lB & 15);
    int k = tk * 32 + ((lB >> 4) & 1) * 8 + (pos & 7) + ((pos >> 3) & 1) * 16;
    int gsel = n >> 10, j = n & 1023;
    const float* src = (gsel == 0) ? wi : (gsel == 1) ? wf : (gsel == 2) ? wg : wo;
    Bpack[i] = f2bf(src[(size_t)k * HID + j]);
}

__global__ __launch_bounds__(256)
void init_state(unsigned short* __restrict__ Hpack, float* __restrict__ c) {
    int i = blockIdx.x * 256 + threadIdx.x;              // BATCH*HID threads
    Hpack[i] = 0;
    c[i] = 0.0f;
}

// ---------------- host launcher -------------------------------------------
extern "C" void kernel_launch(void* const* d_in, const int* in_sizes, int n_in,
                              void* d_out, int out_size, void* d_ws, size_t ws_size,
                              hipStream_t stream) {
    const float* x    = (const float*)d_in[0];
    const float* W_ii = (const float*)d_in[1];
    const float* W_hi = (const float*)d_in[2];
    const float* W_if = (const float*)d_in[3];
    const float* W_hf = (const float*)d_in[4];
    const float* W_ig = (const float*)d_in[5];
    const float* W_hg = (const float*)d_in[6];
    const float* W_io = (const float*)d_in[7];
    const float* W_ho = (const float*)d_in[8];
    const float* b_ii = (const float*)d_in[9];
    const float* b_hi = (const float*)d_in[10];
    const float* b_if = (const float*)d_in[11];
    const float* b_hf = (const float*)d_in[12];
    const float* b_ig = (const float*)d_in[13];
    const float* b_hg = (const float*)d_in[14];
    const float* b_io = (const float*)d_in[15];
    const float* b_ho = (const float*)d_in[16];
    float* out = (float*)d_out;

    // ---- workspace carve-up (all offsets 256B aligned) ----
    char* ws = (char*)d_ws;
    size_t off = 0;
    unsigned short* Xpack = (unsigned short*)(ws + off); off += (size_t)BS * INF_ * 2;   // 128 MB
    unsigned short* WiP   = (unsigned short*)(ws + off); off += (size_t)INF_ * G4 * 2;   //   8 MB
    unsigned short* WhP   = (unsigned short*)(ws + off); off += (size_t)HID * G4 * 2;    //   8 MB
    unsigned short* gxT   = (unsigned short*)(ws + off); off += (size_t)BS * G4 * 2;     // 512 MB
    float*          gates = (float*)(ws + off);          off += (size_t)BATCH * G4 * 4;  //   4 MB
    unsigned short* Hpack = (unsigned short*)(ws + off); off += (size_t)BATCH * HID * 2; // 0.5 MB
    float*          Cst   = (float*)(ws + off);          off += (size_t)BATCH * HID * 4; //   1 MB
    (void)ws_size; (void)n_in; (void)in_sizes; (void)out_size;

    // ---- prep / packing ----
    pack_x<<<(BS * INF_) / 256, 256, 0, stream>>>(x, Xpack);
    pack_w<<<(INF_ * G4) / 256, 256, 0, stream>>>(W_ii, W_if, W_ig, W_io, WiP);
    pack_w<<<(HID * G4) / 256, 256, 0, stream>>>(W_hi, W_hf, W_hg, W_ho, WhP);
    init_state<<<(BATCH * HID) / 256, 256, 0, stream>>>(Hpack, Cst);

    // ---- big parallel GEMM: all x-projections ----
    lstm_gemm_x<<<dim3(BS / 256, G4 / 64), 256, 0, stream>>>(Xpack, WiP, gxT);

    // ---- sequential recurrence ----
    for (int t = 0; t < SEQ; ++t) {
        const unsigned short* gx_t = gxT + (size_t)t * BATCH * G4;
        lstm_gemm_h<<<dim3(BATCH / 128, G4 / 64), 256, 0, stream>>>(Hpack, WhP, gx_t, gates);
        lstm_cell<<<(BATCH * HID) / 256, 256, 0, stream>>>(
            gates, Cst, Hpack, b_ii, b_hi, b_if, b_hf, b_ig, b_hg, b_io, b_ho,
            out, (t == SEQ - 1) ? 1 : 0);
    }
}
